// MulHeadAttentionDecoder_6390911337073
// MI455X (gfx1250) — matrix operand order repaired and checked by reading, not compiled, not run on hardware
//
#include <hip/hip_runtime.h>

typedef _Float16 half_t;
typedef __attribute__((ext_vector_type(16))) _Float16 v16h;
typedef __attribute__((ext_vector_type(8)))  _Float16 v8h;
typedef __attribute__((ext_vector_type(8)))  float    v8f;
typedef __attribute__((ext_vector_type(4)))  int      v4i;

#define LN_EPS 1e-5f
#define DHEAD 128

#if defined(__has_builtin)
#  if __has_builtin(__builtin_amdgcn_global_load_async_to_lds_b128)
#    define USE_ASYNC_LDS 1
#  endif
#endif
#ifndef USE_ASYNC_LDS
#  define USE_ASYNC_LDS 0
#endif

// ---------------------------------------------------------------------------
// WMMA helpers (CDNA5 wave32, V_WMMA_F32_16X16X32_F16)
// ---------------------------------------------------------------------------
__device__ __forceinline__ v8f wmma16(v16h a, v16h b, v8f c) {
  // (neg_a, A, neg_b, B, c_mod, C, reuse_a, reuse_b)
  return __builtin_amdgcn_wmma_f32_16x16x32_f16(false, a, false, b, (short)0, c, false, false);
}

// A-matrix 16x32 f16 fragment. Per ISA layout:
// lanes 0-15 : row M=lane, K = k0+[0..7] (VGPR0-3) and k0+[16..23] (VGPR4-7)
// lanes 16-31: row M=lane-16, K = k0+[8..15] and k0+[24..31]
__device__ __forceinline__ v16h load_a_frag(const half_t* base, int ld,
                                            int row, int k0, int lane) {
  int koff = (lane >> 4) * 8;
  const half_t* p = base + (size_t)row * ld + k0 + koff;
  v8h lo = *(const v8h*)(p);
  v8h hi = *(const v8h*)(p + 16);
  v16h r;
#pragma unroll
  for (int i = 0; i < 8; ++i) { r[i] = lo[i]; r[i + 8] = hi[i]; }
  return r;
}

// B-matrix 32x16 f16 fragment, sourced from a row-major [N,K] matrix
// (i.e. B[k][n] = M[n][k]):
// lanes 0-15 : column n=lane, K = k0+[0..7] and k0+[16..23]
// lanes 16-31: column n=lane-16, K = k0+[8..15] and k0+[24..31]
__device__ __forceinline__ v16h load_b_frag(const half_t* base, int ld,
                                            int col0, int k0, int lane) {
  int col = col0 + (lane & 15);
  int koff = (lane >> 4) * 8;
  const half_t* p = base + (size_t)col * ld + k0 + koff;
  v8h lo = *(const v8h*)(p);
  v8h hi = *(const v8h*)(p + 16);
  v16h r;
#pragma unroll
  for (int i = 0; i < 8; ++i) { r[i] = lo[i]; r[i + 8] = hi[i]; }
  return r;
}

// 16-byte global -> LDS copy: async DMA path if available, else through VGPRs.
__device__ __forceinline__ void copy16_to_lds(const half_t* g, half_t* l) {
#if USE_ASYNC_LDS
  __builtin_amdgcn_global_load_async_to_lds_b128(
      (__attribute__((address_space(1))) v4i*)g,
      (__attribute__((address_space(3))) v4i*)l, 0, 0);
#else
  *(v8h*)l = *(const v8h*)g;
#endif
}

__device__ __forceinline__ void wait_async_tail() {
#if USE_ASYNC_LDS
  __asm__ volatile("s_wait_asynccnt 0x4" ::: "memory");  // keep next stage in flight
#endif
}
__device__ __forceinline__ void wait_async_all() {
#if USE_ASYNC_LDS
  __asm__ volatile("s_wait_asynccnt 0x0" ::: "memory");
#endif
}

// ---------------------------------------------------------------------------
// Generic GEMM: C[M,N] = A[M,K](f16) @ W[N,K](f16)^T + bias, opt ReLU.
// Block tile 128x128 (8 waves, wave tile 64x32). K staged through LDS in
// double-buffered 128x32 tiles via async global->LDS copies (ASYNCcnt).
// grid = (N/128, M/128), block = 256.
// ---------------------------------------------------------------------------
__global__ __launch_bounds__(256) void gemm_kernel(
    const half_t* __restrict__ A, int lda,
    const half_t* __restrict__ W, int ldw,
    const float* __restrict__ bias,
    float* __restrict__ Cf, half_t* __restrict__ Ch, int ldc,
    int K, int relu)
{
  __shared__ __align__(16) half_t ldsA[2][128 * 32];
  __shared__ __align__(16) half_t ldsB[2][128 * 32];

  int tid  = threadIdx.x;
  int lane = tid & 31;
  int wid  = tid >> 5;
  int bm0  = blockIdx.y * 128;
  int bn0  = blockIdx.x * 128;
  int wm   = (wid & 1) * 64;
  int wn   = (wid >> 1) * 32;
  int nk   = K >> 5;

  // stage one 128x32 A tile + 128x32 B tile; 512 chunks of 16B each per tile,
  // 256 threads -> 2 A-chunks + 2 B-chunks per thread (4 async loads).
  auto stage = [&](int buf, int k0) {
    int chunk = tid;
#pragma unroll
    for (int rep = 0; rep < 2; ++rep, chunk += 256) {
      int row = chunk >> 2;
      int c8  = (chunk & 3) * 8;
      copy16_to_lds(A + (size_t)(bm0 + row) * lda + k0 + c8, &ldsA[buf][row * 32 + c8]);
      copy16_to_lds(W + (size_t)(bn0 + row) * ldw + k0 + c8, &ldsB[buf][row * 32 + c8]);
    }
  };

  stage(0, 0);
  v8f acc[4][2] = {};
  for (int kt = 0; kt < nk; ++kt) {
    int cur = kt & 1;
    bool more = (kt + 1) < nk;
    if (more) stage(1 - cur, (kt + 1) * 32);
    if (more) wait_async_tail(); else wait_async_all();
    __syncthreads();  // all waves certified their slice of buffer `cur`

    v16h a[4], b[2];
#pragma unroll
    for (int mi = 0; mi < 4; ++mi)
      a[mi] = load_a_frag(&ldsA[cur][0], 32, wm + mi * 16 + (lane & 15), 0, lane);
#pragma unroll
    for (int ni = 0; ni < 2; ++ni)
      b[ni] = load_b_frag(&ldsB[cur][0], 32, wn + ni * 16, 0, lane);
#pragma unroll
    for (int mi = 0; mi < 4; ++mi)
#pragma unroll
      for (int ni = 0; ni < 2; ++ni)
        acc[mi][ni] = wmma16(a[mi], b[ni], acc[mi][ni]);

    __syncthreads();  // done reading `cur` before it is re-staged
  }

  // C layout: lane l holds column N = l%16, rows M = v + 8*(l/16)
  int rofs = (lane >> 4) * 8;
  int cl   = lane & 15;
#pragma unroll
  for (int ni = 0; ni < 2; ++ni) {
    int col = bn0 + wn + ni * 16 + cl;
    float bv = bias ? bias[col] : 0.f;
#pragma unroll
    for (int mi = 0; mi < 4; ++mi) {
#pragma unroll
      for (int v = 0; v < 8; ++v) {
        float x = acc[mi][ni][v] + bv;
        if (relu) x = fmaxf(x, 0.f);
        int row = bm0 + wm + mi * 16 + rofs + v;
        if (Cf) Cf[(size_t)row * ldc + col] = x;
        if (Ch) Ch[(size_t)row * ldc + col] = (half_t)x;
      }
    }
  }
}

// ---------------------------------------------------------------------------
// Flash attention (no mask, full softmax over L keys).
// One wave per (head, 16-row query tile). d_head = 128, NH = 8.
// Q,K: f16 [L,H] row-major; Vt: f16 [H,L] (pre-transposed); Oh: f16 [L,H].
// ---------------------------------------------------------------------------
__global__ __launch_bounds__(128) void flash_kernel(
    const half_t* __restrict__ Q, const half_t* __restrict__ Kh,
    const half_t* __restrict__ Vt, half_t* __restrict__ Oh,
    int L, int H, float scale)
{
  __shared__ __align__(16) half_t plds[4 * 16 * 32];  // per-wave 16x32 f16 P tile
  int lane = threadIdx.x & 31;
  int w    = threadIdx.x >> 5;
  int gw   = blockIdx.x * 4 + w;
  int mtiles = L >> 4;
  int h  = gw / mtiles;
  int mt = gw % mtiles;
  half_t* pw = plds + w * (16 * 32);

  const half_t* Qh0 = Q + h * DHEAD;
  const half_t* Kh0 = Kh + h * DHEAD;

  // Pre-scale Q fragments by 1/sqrt(d).
  v16h qa[4];
  half_t hs = (half_t)scale;
#pragma unroll
  for (int kk = 0; kk < 4; ++kk) {
    qa[kk] = load_a_frag(Qh0, H, mt * 16 + (lane & 15), kk * 32, lane);
#pragma unroll
    for (int i = 0; i < 16; ++i) qa[kk][i] = qa[kk][i] * hs;
  }

  v8f acc[8] = {};
  float m_i[8], l_i[8];
#pragma unroll
  for (int v = 0; v < 8; ++v) { m_i[v] = -1e30f; l_i[v] = 0.f; }

  int prow = (lane >> 4) * 8;
  int pcol = lane & 15;

  for (int kb = 0; kb < (L >> 5); ++kb) {
    // scores for 32 keys: two 16x16 tiles, K-loop over d=128
    v8f s0 = {}, s1 = {};
#pragma unroll
    for (int kk = 0; kk < 4; ++kk) {
      v16h b0 = load_b_frag(Kh0, H, kb * 32,      kk * 32, lane);
      v16h b1 = load_b_frag(Kh0, H, kb * 32 + 16, kk * 32, lane);
      s0 = wmma16(qa[kk], b0, s0);
      s1 = wmma16(qa[kk], b1, s1);
    }

    // online softmax; row stats reduced across the 16-lane half-wave
    float r[8];
#pragma unroll
    for (int v = 0; v < 8; ++v) {
      float t = fmaxf(s0[v], s1[v]);
#pragma unroll
      for (int off = 1; off < 16; off <<= 1) t = fmaxf(t, __shfl_xor(t, off, 16));
      float mnew = fmaxf(m_i[v], t);
      r[v] = __expf(m_i[v] - mnew);
      float p0 = __expf(s0[v] - mnew);
      float p1 = __expf(s1[v] - mnew);
      s0[v] = p0; s1[v] = p1;
      float ps = p0 + p1;
#pragma unroll
      for (int off = 1; off < 16; off <<= 1) ps += __shfl_xor(ps, off, 16);
      l_i[v] = l_i[v] * r[v] + ps;
      m_i[v] = mnew;
    }
#pragma unroll
    for (int dv = 0; dv < 8; ++dv)
#pragma unroll
      for (int v = 0; v < 8; ++v) acc[dv][v] *= r[v];

    // C-layout P -> LDS (f16) -> reload as A fragment (K=32)
#pragma unroll
    for (int v = 0; v < 8; ++v) {
      pw[(prow + v) * 32 + pcol]      = (half_t)s0[v];
      pw[(prow + v) * 32 + pcol + 16] = (half_t)s1[v];
    }
    __asm__ volatile("s_wait_dscnt 0x0" ::: "memory");
    v16h pa = load_a_frag(pw, 32, lane & 15, 0, lane);

    // O += P @ V : B fragments from Vt rows (contiguous keys)
#pragma unroll
    for (int dv = 0; dv < 8; ++dv) {
      v16h bv = load_b_frag(Vt, L, h * DHEAD + dv * 16, kb * 32, lane);
      acc[dv] = wmma16(pa, bv, acc[dv]);
    }
  }

  // normalize and store f16 output
#pragma unroll
  for (int v = 0; v < 8; ++v) {
    float inv = 1.0f / l_i[v];
    int row = mt * 16 + prow + v;
#pragma unroll
    for (int dv = 0; dv < 8; ++dv)
      Oh[(size_t)row * H + h * DHEAD + dv * 16 + pcol] = (half_t)(acc[dv][v] * inv);
  }
}

// ---------------------------------------------------------------------------
// Residual add + LayerNorm (H = 1024, 4 elems/thread, 256 threads/row).
// ---------------------------------------------------------------------------
__global__ __launch_bounds__(256) void add_ln_kernel(
    const float* __restrict__ Y, const float* __restrict__ X,
    const float* __restrict__ g, const float* __restrict__ b,
    float* __restrict__ Of, half_t* __restrict__ Oh, int H)
{
  int row = blockIdx.x;
  int tid = threadIdx.x;
  __shared__ float red[256];
  float z[4];
  float s = 0.f;
#pragma unroll
  for (int i = 0; i < 4; ++i) {
    int c = i * 256 + tid;
    z[i] = Y[(size_t)row * H + c] + X[(size_t)row * H + c];
    s += z[i];
  }
  red[tid] = s; __syncthreads();
  for (int st = 128; st > 0; st >>= 1) { if (tid < st) red[tid] += red[tid + st]; __syncthreads(); }
  float mu = red[0] / H;
  __syncthreads();
  float vs = 0.f;
#pragma unroll
  for (int i = 0; i < 4; ++i) { float d = z[i] - mu; vs += d * d; }
  red[tid] = vs; __syncthreads();
  for (int st = 128; st > 0; st >>= 1) { if (tid < st) red[tid] += red[tid + st]; __syncthreads(); }
  float rstd = rsqrtf(red[0] / H + LN_EPS);
#pragma unroll
  for (int i = 0; i < 4; ++i) {
    int c = i * 256 + tid;
    float o = g[c] * (z[i] - mu) * rstd + b[c];
    Of[(size_t)row * H + c] = o;
    if (Oh) Oh[(size_t)row * H + c] = (half_t)o;
  }
}

// ---------------------------------------------------------------------------
// fp32 -> f16 convert with optional K padding (zeros beyond cols).
// ---------------------------------------------------------------------------
__global__ void cvt_f16_pad_kernel(const float* __restrict__ src, half_t* __restrict__ dst,
                                   int rows, int cols, int ldd) {
  int idx = blockIdx.x * 256 + threadIdx.x;
  if (idx >= rows * ldd) return;
  int r = idx / ldd, c = idx - r * ldd;
  dst[idx] = (c < cols) ? (half_t)src[(size_t)r * cols + c] : (half_t)0.f;
}

__global__ void transpose_f16_kernel(const half_t* __restrict__ src, half_t* __restrict__ dst,
                                     int R, int C) {
  int idx = blockIdx.x * 256 + threadIdx.x;
  if (idx >= R * C) return;
  int r = idx / C, c = idx - r * C;
  dst[(size_t)c * R + r] = src[idx];
}

// ---------------------------------------------------------------------------
// Host-side orchestration
// ---------------------------------------------------------------------------
extern "C" void kernel_launch(void* const* d_in, const int* in_sizes, int n_in,
                              void* d_out, int out_size, void* d_ws, size_t ws_size,
                              hipStream_t stream) {
  const int Nn = 2048, H = 1024, DE = 300, DEP = 320, FF = 2048;

  const float* IIN = (const float*)d_in[0];
  const float* WIN = (const float*)d_in[1];
  const float* SIN = (const float*)d_in[2];
  const float* WWD = (const float*)d_in[3];
  const float* BWD = (const float*)d_in[4];
  const float* WQKV[3] = {(const float*)d_in[5], (const float*)d_in[9],  (const float*)d_in[13]};
  const float* BQKV[3] = {(const float*)d_in[6], (const float*)d_in[10], (const float*)d_in[14]};
  const float* WO[3]   = {(const float*)d_in[7], (const float*)d_in[11], (const float*)d_in[15]};
  const float* BO[3]   = {(const float*)d_in[8], (const float*)d_in[12], (const float*)d_in[16]};
  const float *W1a[3], *B1a[3], *W2a[3], *B2a[3], *Ga[3], *Ba[3];
  for (int j = 0; j < 3; ++j) {
    const int base = 17 + 6 * j;
    W1a[j] = (const float*)d_in[base + 0];
    B1a[j] = (const float*)d_in[base + 1];
    W2a[j] = (const float*)d_in[base + 2];
    B2a[j] = (const float*)d_in[base + 3];
    Ga[j]  = (const float*)d_in[base + 4];
    Ba[j]  = (const float*)d_in[base + 5];
  }

  // workspace bump allocator
  char* wsb = (char*)d_ws;
  size_t off = 0;
  auto alloc = [&](size_t bytes) -> void* {
    void* p = wsb + off;
    off += (bytes + 255) & ~(size_t)255;
    return p;
  };

  half_t* wwd_h = (half_t*)alloc((size_t)H * DEP * 2);
  half_t *wqkv_h[3], *wo_h[3], *w1_h[3], *w2_h[3];
  for (int j = 0; j < 3; ++j) {
    wqkv_h[j] = (half_t*)alloc((size_t)3 * H * H * 2);
    wo_h[j]   = (half_t*)alloc((size_t)H * H * 2);
    w1_h[j]   = (half_t*)alloc((size_t)FF * H * 2);
    w2_h[j]   = (half_t*)alloc((size_t)H * FF * 2);
  }
  half_t* wpad  = (half_t*)alloc((size_t)Nn * DEP * 2);
  half_t* ipad  = (half_t*)alloc((size_t)Nn * DEP * 2);
  half_t* s_h   = (half_t*)alloc((size_t)Nn * H * 2);
  half_t* w_h   = (half_t*)alloc((size_t)Nn * H * 2);
  half_t* i_h   = (half_t*)alloc((size_t)Nn * H * 2);
  half_t* qp    = (half_t*)alloc((size_t)Nn * H * 2);
  half_t* kp    = (half_t*)alloc((size_t)Nn * H * 2);
  half_t* vp    = (half_t*)alloc((size_t)Nn * H * 2);
  half_t* vt    = (half_t*)alloc((size_t)H * Nn * 2);
  half_t* oh    = (half_t*)alloc((size_t)Nn * H * 2);
  half_t* attn_h= (half_t*)alloc((size_t)Nn * H * 2);
  half_t* iw_h  = (half_t*)alloc((size_t)Nn * H * 2);
  half_t* sw_h  = (half_t*)alloc((size_t)Nn * H * 2);
  half_t* ffn1  = (half_t*)alloc((size_t)Nn * FF * 2);
  float*  attn_f= (float*)alloc((size_t)Nn * H * 4);
  float*  ffn2  = (float*)alloc((size_t)Nn * H * 4);
  float*  iw_f  = (float*)alloc((size_t)Nn * H * 4);
  float*  sw_f  = (float*)alloc((size_t)Nn * H * 4);

  auto cvt = [&](const float* src, half_t* dst, int rows, int cols, int ldd) {
    int n = rows * ldd;
    cvt_f16_pad_kernel<<<(n + 255) / 256, 256, 0, stream>>>(src, dst, rows, cols, ldd);
  };
  auto gemm = [&](const half_t* A, int lda, const half_t* Wm, int ldw, const float* bias,
                  float* Cf, half_t* Chx, int ldc, int Ncols, int K, int relu) {
    dim3 g(Ncols / 128, Nn / 128), bl(256);
    gemm_kernel<<<g, bl, 0, stream>>>(A, lda, Wm, ldw, bias, Cf, Chx, ldc, K, relu);
  };

  // --- convert weights & inputs to f16 (weights L2-resident afterwards) ---
  cvt(WWD, wwd_h, H, DE, DEP);
  for (int j = 0; j < 3; ++j) {
    cvt(WQKV[j], wqkv_h[j], 3 * H, H, H);
    cvt(WO[j],   wo_h[j],   H, H, H);
    cvt(W1a[j],  w1_h[j],   FF, H, H);
    cvt(W2a[j],  w2_h[j],   H, FF, FF);
  }
  cvt(WIN, wpad, Nn, DE, DEP);
  cvt(IIN, ipad, Nn, DE, DEP);
  cvt(SIN, s_h,  Nn, H, H);

  // --- embeddings: w = winpute@Wwd^T + bwd ; i = iinpute@Wwd^T + bwd ---
  gemm(wpad, DEP, wwd_h, DEP, BWD, nullptr, w_h, H, H, DEP, 0);
  gemm(ipad, DEP, wwd_h, DEP, BWD, nullptr, i_h, H, H, DEP, 0);

  auto mha = [&](const half_t* q, const half_t* k, const half_t* v, int j) {
    gemm(q, H, wqkv_h[j],                    H, BQKV[j],         nullptr, qp, H, H, H, 0);
    gemm(k, H, wqkv_h[j] + (size_t)H * H,    H, BQKV[j] + H,     nullptr, kp, H, H, H, 0);
    gemm(v, H, wqkv_h[j] + (size_t)2 * H * H,H, BQKV[j] + 2 * H, nullptr, vp, H, H, H, 0);
    int n = Nn * H;
    transpose_f16_kernel<<<(n + 255) / 256, 256, 0, stream>>>(vp, vt, Nn, H);
    flash_kernel<<<(8 * (Nn / 16)) / 4, 128, 0, stream>>>(qp, kp, vt, oh, Nn, H,
                                                          0.08838834764831845f);
    gemm(oh, H, wo_h[j], H, BO[j], attn_f, attn_h, H, H, H, 0);
  };
  auto ffn = [&](int j, float* outF, half_t* outH) {
    gemm(attn_h, H, w1_h[j], H,  B1a[j], nullptr, ffn1, FF, FF, H, 1);
    gemm(ffn1,  FF, w2_h[j], FF, B2a[j], ffn2, nullptr, H, H, FF, 0);
    add_ln_kernel<<<Nn, 256, 0, stream>>>(ffn2, attn_f, Ga[j], Ba[j], outF, outH, H);
  };

  // iw = ffn1(mha_ig(w, i, i))
  mha(w_h, i_h, i_h, 0);
  ffn(0, iw_f, iw_h);
  // sw = ffn2(mha_h1(w, s, s))
  mha(w_h, s_h, s_h, 1);
  ffn(1, sw_f, sw_h);
  // out = ffn3(mha_h2(s, iw, sw))
  mha(s_h, iw_h, sw_h, 2);
  ffn(2, (float*)d_out, nullptr);
}
